// MultiHeadedGlobalAttention_57741540327433
// MI455X (gfx1250) — matrix operand order repaired and checked by reading, not compiled, not run on hardware
//
#include <hip/hip_runtime.h>
#include <math.h>

// Problem constants (from the reference)
#define B_   2
#define S_   2048
#define G_   16
#define H_   8
#define E_   64      // Ek == Ev
#define DM   512     // d_model == kdim == vdim
#define NC_  32      // number of S-chunks for the parallel causal scan
#define CS_  64      // chunk size (NC_*CS_ == S_)

typedef __attribute__((ext_vector_type(16))) __bf16    v16bf;
typedef __attribute__((ext_vector_type(8)))  float     v8f;
typedef __attribute__((ext_vector_type(4)))  unsigned  v4u;
typedef __attribute__((ext_vector_type(8)))  unsigned  v8u;

// ---------- bf16 helpers: native HW convert (v_cvt), then pack ----------
__device__ __forceinline__ unsigned bf16pack(float lo, float hi) {
  unsigned short a = __builtin_bit_cast(unsigned short, (__bf16)lo);
  unsigned short b = __builtin_bit_cast(unsigned short, (__bf16)hi);
  return (unsigned)a | ((unsigned)b << 16);
}

// Fragment-order mapping (inverse of the ISA 16-bit A 16x32 striping):
//   element e = (k&7) + (k>=16 ? 8 : 0),  lane-half = (k>>3)&1
// B 32x16 striping: same as A (element e = (k&7) + (k>=16 ? 8 : 0), lane-half = (k>>3)&1)

// =====================================================================
// Kernel 1: Y(MxDM) = X(MxDM) @ W(DMxDM); LDS panels stored in WMMA
// fragment order so each fragment is two ds_load_b128s. GUARD only for
// the ragged context GEMM (M=32); the big GEMMs are branch-free.
// =====================================================================
template <bool GUARD>
__global__ __launch_bounds__(256) void gemm512_wmma(const float* __restrict__ X,
                                                    const float* __restrict__ W,
                                                    float* __restrict__ Y, int M) {
  __shared__ __align__(32) unsigned short Af[4][32][16];  // [mi][lane][e]
  __shared__ __align__(32) unsigned short Bf[4][32][16];  // [ni][lane][e]
  const int t = threadIdx.x;
  const int lane = t & 31, wave = t >> 5;
  const int lhalf = lane >> 4, lm = lane & 15;
  const int m0 = blockIdx.x * 64, n0 = blockIdx.y * 64;
  const int mi = wave & 3;          // M sub-tile 0..3
  const int ng = wave >> 2;         // N pair 0..1
  v8f acc0 = {}, acc1 = {};
  for (int k0 = 0; k0 < DM; k0 += 32) {
    if (k0 + 32 < DM) {
      int pr = m0 + (t & 63);
      if (!GUARD || pr < M) __builtin_prefetch(&X[pr * DM + k0 + 32], 0, 1);
    }
#pragma unroll
    for (int j = 0; j < 4; ++j) {
      int p = t + 256 * j;                  // 0..1023 (pair index)
      // ---- A panel 64x32: pair along k (contiguous global float2) ----
      int ar = p >> 4;                      // row 0..63
      int kp = (p & 15) * 2;                // even k
      int gr = m0 + ar;
      float x0 = 0.f, x1 = 0.f;
      if (!GUARD || gr < M) {
        const float* sp = &X[gr * DM + k0 + kp];
        x0 = sp[0]; x1 = sp[1];
      }
      int ae = (kp & 7) + (kp >= 16 ? 8 : 0);
      int al = (ar & 15) + (((kp >> 3) & 1) << 4);
      *(unsigned*)&Af[ar >> 4][al][ae] = bf16pack(x0, x1);
      // ---- B panel 32x64: pair along k (strided global) ----
      int n  = p & 63;
      int kq = (p >> 6) * 2;                // even k 0..30
      float w0 = W[(k0 + kq) * DM + n0 + n];
      float w1 = W[(k0 + kq + 1) * DM + n0 + n];
      int be = (kq & 7) + (kq >= 16 ? 8 : 0);
      int bl = (n & 15) + (((kq >> 3) & 1) << 4);
      *(unsigned*)&Bf[n >> 4][bl][be] = bf16pack(w0, w1);
    }
    __syncthreads();
    v16bf a  = *(const v16bf*)&Af[mi][lane][0];
    v16bf b0 = *(const v16bf*)&Bf[ng * 2][lane][0];
    v16bf b1 = *(const v16bf*)&Bf[ng * 2 + 1][lane][0];
    acc0 = __builtin_amdgcn_wmma_f32_16x16x32_bf16(false, a, false, b0, (short)0, acc0, false, false);
    acc1 = __builtin_amdgcn_wmma_f32_16x16x32_bf16(false, a, false, b1, (short)0, acc1, false, false);
    __syncthreads();
  }
#pragma unroll
  for (int r = 0; r < 8; ++r) {             // C: M = r + 8*half, N = lm
    int grow = m0 + mi * 16 + lhalf * 8 + r;
    if (!GUARD || grow < M) {
      Y[grow * DM + n0 + ng * 32 + lm]      = acc0[r];
      Y[grow * DM + n0 + ng * 32 + 16 + lm] = acc1[r];
    }
  }
}

// =====================================================================
// Kernel 2: raw logits w[b,h,g,s] = (cs . kk)/8 via WMMA; block handles
// (b,h) and 8 consecutive s-tiles (one per wave); LDS fragment staging.
// =====================================================================
__global__ __launch_bounds__(256) void wraw_wmma(const float* __restrict__ csM,  // (B*G,512)
                                                 const float* __restrict__ kkM,  // (B*S,512)
                                                 float* __restrict__ wbuf,
                                                 float* __restrict__ gsum) {
  __shared__ __align__(32) unsigned short Acs[2][32][16];     // cs 16x64 (2 k-steps)
  __shared__ __align__(32) unsigned short Bkk[8][2][32][16];  // per-wave kk tiles
  __shared__ float red[256];
  const int t = threadIdx.x, lane = t & 31, wave = t >> 5;
  const int lhalf = lane >> 4, lm = lane & 15;
  int stgrp = blockIdx.x & 15;
  int h = (blockIdx.x >> 4) & 7;
  int b = blockIdx.x >> 7;
  // ---- stage A (cs): 512 pairs ----
#pragma unroll
  for (int j = 0; j < 2; ++j) {
    int p = t + 256 * j;                    // 0..511
    int g  = p >> 5;
    int fp = (p & 31) * 2;                  // even feature 0..62
    const float* sp = &csM[(b * G_ + g) * DM + h * E_ + fp];
    int kl = fp & 31;
    int ae = (kl & 7) + (kl >= 16 ? 8 : 0);
    int al = g + (((kl >> 3) & 1) << 4);
    *(unsigned*)&Acs[fp >> 5][al][ae] = bf16pack(sp[0], sp[1]);
  }
  // ---- stage B (kk) for 8 s-tiles: 4096 pairs ----
#pragma unroll
  for (int j = 0; j < 16; ++j) {
    int p = t + 256 * j;                    // 0..4095
    int w = p >> 9;                         // s-tile 0..7
    int r = p & 511;
    int n  = r & 15;                        // s within tile
    int kp = (r >> 4) * 2;                  // even feature 0..62
    const float* sp = &kkM[(b * S_ + (stgrp * 8 + w) * 16 + n) * DM + h * E_ + kp];
    int kl = kp & 31;
    int be = (kl & 7) + (kl >= 16 ? 8 : 0);
    int bl = n + (((kl >> 3) & 1) << 4);
    *(unsigned*)&Bkk[w][kp >> 5][bl][be] = bf16pack(sp[0], sp[1]);
  }
  __syncthreads();
  v8f acc = {};
#pragma unroll
  for (int ks = 0; ks < 2; ++ks) {
    v16bf a  = *(const v16bf*)&Acs[ks][lane][0];
    v16bf bb = *(const v16bf*)&Bkk[wave][ks][lane][0];
    acc = __builtin_amdgcn_wmma_f32_16x16x32_bf16(false, a, false, bb, (short)0, acc, false, false);
  }
  float lsum = 0.f;
#pragma unroll
  for (int r = 0; r < 8; ++r) {
    int g = lhalf * 8 + r;
    int s = (stgrp * 8 + wave) * 16 + lm;
    float v = acc[r] * 0.125f;              // 1/sqrt(64)
    wbuf[(((b * H_ + h) * G_ + g) * S_) + s] = v;
    lsum += v;
  }
  red[t] = lsum;
  __syncthreads();
  for (int off = 128; off; off >>= 1) {
    if (t < off) red[t] += red[t + off];
    __syncthreads();
  }
  if (t == 0) atomicAdd(gsum, red[0]);
}

__global__ void zero_scalar(float* p) { p[0] = 0.f; }

__global__ void exp_kernel(float* __restrict__ wbuf, const float* __restrict__ gsum) {
  const int N = B_ * H_ * G_ * S_;
  int i = blockIdx.x * blockDim.x + threadIdx.x;
  float mean = gsum[0] * (1.0f / (float)N);
  if (i < N) wbuf[i] = __expf(wbuf[i] - mean);
}

// =====================================================================
// Kernel 4: per-chunk totals sPk/sPv = w_chunk(16xCS) @ {k,v}_chunk(CSx64)
// via WMMA with LDS fragment staging; waves 0-3 -> Pk, 4-7 -> Pv.
// =====================================================================
__global__ __launch_bounds__(256) void chunk_sums_wmma(const float* __restrict__ wbuf,
                                                       const float* __restrict__ kkM,
                                                       const float* __restrict__ vsM,
                                                       float* __restrict__ sPk,
                                                       float* __restrict__ sPv,
                                                       float* __restrict__ ssw) {
  __shared__ __align__(32) unsigned short Aw[2][32][16];     // w 16xCS
  __shared__ __align__(32) unsigned short Bm[8][2][32][16];  // per-wave {k,v} tiles
  const int t = threadIdx.x, lane = t & 31, wave = t >> 5;
  const int lhalf = lane >> 4, lm = lane & 15;
  int c  = blockIdx.x % NC_;
  int bh = blockIdx.x / NC_;
  int h = bh % H_, b = bh / H_;
  int c0 = c * CS_;
  // ---- stage A (w chunk): 512 pairs, contiguous in wbuf ----
#pragma unroll
  for (int j = 0; j < 2; ++j) {
    int p = t + 256 * j;
    int g  = p >> 5;
    int sp2 = (p & 31) * 2;                  // even s' 0..62
    const float* sp = &wbuf[((bh * G_ + g) * S_) + c0 + sp2];
    int kl = sp2 & 31;
    int ae = (kl & 7) + (kl >= 16 ? 8 : 0);
    int al = g + (((kl >> 3) & 1) << 4);
    *(unsigned*)&Aw[sp2 >> 5][al][ae] = bf16pack(sp[0], sp[1]);
  }
  // ---- stage B ({kk|vs} chunk, 8 tiles): 4096 pairs ----
#pragma unroll
  for (int j = 0; j < 16; ++j) {
    int p = t + 256 * j;
    int w = p >> 9;                          // tile 0..7
    const float* srcp = (w < 4) ? kkM : vsM;
    int ni = w & 3;
    int r = p & 511;
    int n  = r & 15;                         // e-column within tile
    int kp = (r >> 4) * 2;                   // even s' 0..62
    float w0 = srcp[(b * S_ + c0 + kp) * DM + h * E_ + ni * 16 + n];
    float w1 = srcp[(b * S_ + c0 + kp + 1) * DM + h * E_ + ni * 16 + n];
    int kl = kp & 31;
    int be = (kl & 7) + (kl >= 16 ? 8 : 0);
    int bl = n + (((kl >> 3) & 1) << 4);
    *(unsigned*)&Bm[w][kp >> 5][bl][be] = bf16pack(w0, w1);
  }
  __syncthreads();
  v8f acc = {};
#pragma unroll
  for (int ks = 0; ks < 2; ++ks) {
    v16bf a  = *(const v16bf*)&Aw[ks][lane][0];
    v16bf bb = *(const v16bf*)&Bm[wave][ks][lane][0];
    acc = __builtin_amdgcn_wmma_f32_16x16x32_bf16(false, a, false, bb, (short)0, acc, false, false);
  }
  float* dst = (wave < 4) ? sPk : sPv;
  int ni = wave & 3;
#pragma unroll
  for (int r = 0; r < 8; ++r) {
    int g = lhalf * 8 + r;
    dst[(((bh * NC_ + c) * G_ + g) * E_) + ni * 16 + lm] = acc[r];
  }
  if (t < G_) {
    float s = 0.f;
    const float* wrow = wbuf + ((bh * G_ + t) * S_) + c0;
    for (int i = 0; i < CS_; ++i) s += wrow[i];
    ssw[(bh * G_ + t) * NC_ + c] = s;
  }
}

// Kernel 5: serial exclusive scan over the 32 chunks (tiny)
__global__ void prefix_kernel(float* __restrict__ sPk, float* __restrict__ sPv,
                              float* __restrict__ ssw) {
  int tid = blockIdx.x * blockDim.x + threadIdx.x;
  if (tid < B_ * H_ * G_ * E_) {
    int e = tid & 63, g = (tid >> 6) & 15, bh = tid >> 10;
    float rk = 0.f, rv = 0.f;
    for (int c = 0; c < NC_; ++c) {
      int idx = (((bh * NC_ + c) * G_ + g) * E_) + e;
      float tk = sPk[idx]; sPk[idx] = rk; rk += tk;
      float tv = sPv[idx]; sPv[idx] = rv; rv += tv;
    }
  }
  if (tid < B_ * H_ * G_) {
    int g = tid & 15, bh = tid >> 4;
    int base = (bh * G_ + g) * NC_;
    float r = 0.f;
    for (int c = 0; c < NC_; ++c) { float tv = ssw[base + c]; ssw[base + c] = r; r += tv; }
  }
}

// =====================================================================
// Kernel 6: fused in-chunk causal scan + softmax-mixture.
// w-tile fetched by the Tensor Data Mover (TDM pad reproduces [16][65]);
// k/v/q chunk copied via float4 (b128) loads/stores.
// =====================================================================
__global__ __launch_bounds__(256) void attn_scan(const float* __restrict__ wbuf,
                                                 const float* __restrict__ kkM,
                                                 const float* __restrict__ vsM,
                                                 const float* __restrict__ qsM,
                                                 const float* __restrict__ sPk,
                                                 const float* __restrict__ sPv,
                                                 const float* __restrict__ ssw,
                                                 float* __restrict__ aout) {
  __shared__ __align__(16) float wc[G_][CS_ + 1];   // TDM destination
  __shared__ __align__(16) float kc[CS_][E_ + 4];   // 16B-aligned rows
  __shared__ __align__(16) float vc[CS_][E_ + 4];
  __shared__ __align__(16) float qc[CS_][E_ + 4];
  __shared__ float Pk[G_][E_ + 1];
  __shared__ float Pv[G_][E_ + 1];
  __shared__ float nrm[G_], logit[G_], cc[G_], oy[E_];
  const int t = threadIdx.x, lane = t & 31, wave = t >> 5;
  int c  = blockIdx.x % NC_;
  int bh = blockIdx.x / NC_;
  int h = bh % H_, b = bh / H_;
  int c0 = c * CS_;

  // ---- TDM: load w tile (rows g=0..15, 64 f32 each, row stride S_) ----
  if (wave == 0) {                       // wave-uniform: only wave 0 issues
    unsigned wc_lds = (unsigned)(unsigned long long)
        (__attribute__((address_space(3))) float*)&wc[0][0];
    unsigned long long ga = (unsigned long long)(wbuf + (size_t)bh * G_ * S_ + c0);
    v4u g0;
    g0.x = 1u;                           // count=1 (valid user descriptor)
    g0.y = wc_lds;                       // lds_addr (bytes)
    g0.z = (unsigned)ga;                 // global_addr[31:0]
    g0.w = (unsigned)(ga >> 32) | (2u << 30);  // global_addr[56:32] | type=2
    v8u g1;
    g1.s0 = (2u << 16)                   // data_size = 4B
          | (1u << 20)                   // pad_enable
          | (5u << 22)                   // pad_interval = 64 DWORDs
          | (0u << 25);                  // pad_amount = 1 DWORD
    g1.s1 = ((unsigned)S_ & 0xFFFFu) << 16;        // tensor_dim0[15:0]
    g1.s2 = ((unsigned)S_ >> 16) | ((unsigned)G_ << 16); // dim0[31:16] | dim1[15:0]
    g1.s3 = ((unsigned)G_ >> 16) | (64u << 16);    // dim1[31:16] | tile_dim0
    g1.s4 = 16u;                                   // tile_dim1 | tile_dim2=0
    g1.s5 = (unsigned)S_;                          // tensor_dim0_stride[31:0]
    g1.s6 = 0u;                                    // stride[47:32] | dim1_stride lo
    g1.s7 = 0u;
    asm volatile("tensor_load_to_lds %0, %1" :: "s"(g0), "s"(g1) : "memory");
    __builtin_amdgcn_s_wait_tensorcnt(0);
  }

  // ---- vectorized preload of k/v/q chunk (b128) and scan state ----
  for (int i = t; i < CS_ * (E_ / 4); i += 256) {
    int sl = i >> 4, e4 = (i & 15) * 4;
    int gidx = (b * S_ + c0 + sl) * DM + h * E_ + e4;
    *(float4*)&kc[sl][e4] = *(const float4*)&kkM[gidx];
    *(float4*)&vc[sl][e4] = *(const float4*)&vsM[gidx];
    *(float4*)&qc[sl][e4] = *(const float4*)&qsM[gidx];
  }
  for (int i = t; i < G_ * E_; i += 256) {
    int g = i >> 6, e = i & 63;
    int idx = (((bh * NC_ + c) * G_ + g) * E_) + e;
    Pk[g][e] = sPk[idx];
    Pv[g][e] = sPv[idx];
  }
  if (t < G_) nrm[t] = ssw[(bh * G_ + t) * NC_ + c];
  __syncthreads();

  const int gq = t >> 6, e64 = t & 63;
  for (int sl = 0; sl < CS_; ++sl) {
    // inclusive causal update of running state
#pragma unroll
    for (int gi = 0; gi < 4; ++gi) {
      int g = gq * 4 + gi;
      float wv = wc[g][sl];
      Pk[g][e64] += wv * kc[sl][e64];
      Pv[g][e64] += wv * vc[sl][e64];
    }
    if (t < G_) nrm[t] += wc[t][sl];
    __syncthreads();
    // logits: wave w owns g = 2w, 2w+1; wave32 shfl_xor tree reduce
    if (t < E_) oy[t] = 0.f;
#pragma unroll
    for (int gi = 0; gi < 2; ++gi) {
      int g = wave * 2 + gi;
      float p = qc[sl][lane] * Pk[g][lane] + qc[sl][lane + 32] * Pk[g][lane + 32];
#pragma unroll
      for (int off = 16; off; off >>= 1) p += __shfl_xor(p, off, 32);
      if (lane == 0) logit[g] = p / fmaxf(nrm[g], 1e-8f);
    }
    __syncthreads();
    // softmax over the 16 groups (lanes 0..15 of wave 0)
    if (t < G_) {
      float v = logit[t];
      float m = v;
#pragma unroll
      for (int off = 8; off; off >>= 1) m = fmaxf(m, __shfl_xor(m, off, 16));
      float ev = __expf(v - m);
      float sm = ev;
#pragma unroll
      for (int off = 8; off; off >>= 1) sm += __shfl_xor(sm, off, 16);
      cc[t] = (ev / sm) / fmaxf(nrm[t], 1e-8f);
    }
    __syncthreads();
    // out[e] = sum_g cc[g] * Pv[g][e] via LDS float atomics (ds_add_f32)
#pragma unroll
    for (int gi = 0; gi < 2; ++gi) {
      int g = wave * 2 + gi;
      float cg = cc[g];
      atomicAdd(&oy[lane],      cg * Pv[g][lane]);
      atomicAdd(&oy[lane + 32], cg * Pv[g][lane + 32]);
    }
    __syncthreads();
    if (t < E_) aout[((b * S_ + c0 + sl) * DM) + h * E_ + t] = oy[t];
    __syncthreads();
  }
}

// =====================================================================
extern "C" void kernel_launch(void* const* d_in, const int* in_sizes, int n_in,
                              void* d_out, int out_size, void* d_ws, size_t ws_size,
                              hipStream_t stream) {
  const float* query   = (const float*)d_in[0];
  const float* key     = (const float*)d_in[1];
  const float* value   = (const float*)d_in[2];
  const float* context = (const float*)d_in[3];
  const float* qproj   = (const float*)d_in[4];
  const float* kproj   = (const float*)d_in[5];
  const float* vproj   = (const float*)d_in[6];
  const float* oproj   = (const float*)d_in[7];
  const float* cproj   = (const float*)d_in[8];
  float* out = (float*)d_out;

  float* ws   = (float*)d_ws;
  const int MBS = B_ * S_;                        // 4096
  float* qs   = ws;
  float* kk   = qs   + MBS * DM;
  float* vs   = kk   + MBS * DM;
  float* cs   = vs   + MBS * DM;
  float* wbuf = cs   + (B_ * G_) * DM;
  float* gsum = wbuf + B_ * H_ * G_ * S_;
  float* sPk  = gsum + 64;
  float* sPv  = sPk  + B_ * H_ * NC_ * G_ * E_;
  float* ssw  = sPv  + B_ * H_ * NC_ * G_ * E_;
  float* aout = ssw  + B_ * H_ * G_ * NC_;

  dim3 blk(256);
  gemm512_wmma<false><<<dim3(MBS / 64, DM / 64), blk, 0, stream>>>(query,   qproj, qs, MBS);
  gemm512_wmma<false><<<dim3(MBS / 64, DM / 64), blk, 0, stream>>>(key,     kproj, kk, MBS);
  gemm512_wmma<false><<<dim3(MBS / 64, DM / 64), blk, 0, stream>>>(value,   vproj, vs, MBS);
  gemm512_wmma<true ><<<dim3(1,        DM / 64), blk, 0, stream>>>(context, cproj, cs, B_ * G_);
  zero_scalar<<<1, 1, 0, stream>>>(gsum);
  wraw_wmma<<<B_ * H_ * 16, blk, 0, stream>>>(cs, kk, wbuf, gsum);
  exp_kernel<<<(B_ * H_ * G_ * S_) / 256, blk, 0, stream>>>(wbuf, gsum);
  chunk_sums_wmma<<<B_ * H_ * NC_, blk, 0, stream>>>(wbuf, kk, vs, sPk, sPv, ssw);
  prefix_kernel<<<(B_ * H_ * G_ * E_) / 256, blk, 0, stream>>>(sPk, sPv, ssw);
  attn_scan<<<B_ * H_ * NC_, blk, 0, stream>>>(wbuf, kk, vs, qs, sPk, sPv, ssw, aout);
  gemm512_wmma<false><<<dim3(MBS / 64, DM / 64), blk, 0, stream>>>(aout, oproj, out, MBS);
}
